// Dcls2d_11957188952682
// MI455X (gfx1250) — matrix operand order repaired and checked by reading, not compiled, not run on hardware
//
#include <hip/hip_runtime.h>

// CDNA5 / gfx1250, wave32. bf16 WMMA (16x16x32, f32 accum) implicit-GEMM conv
// with async global->LDS (ASYNCcnt) double-buffered weight staging.

typedef __attribute__((ext_vector_type(16))) __bf16 v16bf;
typedef __attribute__((ext_vector_type(8)))  float  v8f;

#define N_BATCH 32
#define C_IN    96
#define C_OUT   96
#define HW      64
#define KS      3
#define DIL     7
#define PAD     3
#define NIBLK   3                         // 96 input channels / 32 (WMMA K)
#define NMBLK   6                         // 96 output channels / 16 (WMMA M)
#define FRAG_HALFS 512                    // one A fragment: 32 lanes x 16 bf16
#define BATCH_HALFS (DIL * NMBLK * FRAG_HALFS)   // 21504 halves = 43008 B per (kh,iblk)
#define BATCH_VECS  (BATCH_HALFS / 8)            // 2688 x 16B chunks; 42 per thread
#define NBATCH_K    (DIL * NIBLK)                // 21 K-batches
#define LDSX_STRIDE 104                   // bf16 per input column (52 dwords -> conflict-free)
#define LDSX_W  70                        // 64 pixels + 6 halo

// ---------------------------------------------------------------------------
// Kernel 1: build dilated kernel (bilinear, overlap=add, border=clamp) and
// pack straight into WMMA A-fragment order (bf16).
// A layout (16-bit A 16x32): lane = (o%16) + ((k>>3)&1)*16 ; e = (k>>4)*8 + (k&7)
// Fragment index: ((kh*NIBLK + iblk)*DIL + kw)*NMBLK + m
// ---------------------------------------------------------------------------
__global__ __launch_bounds__(256)
void dcls_pack_kernel(const float* __restrict__ weight,  // [O][I][3][3]
                      const float* __restrict__ P1,      // [3][3][O][I]
                      const float* __restrict__ P2,      // [3][3][O][I]
                      __bf16* __restrict__ AW) {
  int tid = blockIdx.x * 256 + threadIdx.x;
  const int total = DIL * DIL * C_OUT * C_IN;
  if (tid >= total) return;
  int i    = tid % C_IN;
  int o    = (tid / C_IN) % C_OUT;
  int t    = tid / (C_IN * C_OUT);
  int y    = t / DIL;
  int xcol = t % DIL;

  float val = 0.f;
#pragma unroll
  for (int a = 0; a < KS; ++a) {
#pragma unroll
    for (int b = 0; b < KS; ++b) {
      int pidx = ((a * KS + b) * C_OUT + o) * C_IN + i;
      float p1 = P1[pidx] + (float)(DIL / 2);
      float p2 = P2[pidx] + (float)(DIL / 2);
      p1 = fminf(fmaxf(p1, 0.f), (float)(DIL - 1));
      p2 = fminf(fmaxf(p2, 0.f), (float)(DIL - 1));
      float f1 = floorf(p1), f2 = floorf(p2);
      float r1 = p1 - f1,    r2 = p2 - f2;
      int i1 = (int)f1, i2 = (int)f2;
      int i1p = i1 + 1; if (i1p > DIL - 1) i1p = DIL - 1;
      int i2p = i2 + 1; if (i2p > DIL - 1) i2p = DIL - 1;
      float w = weight[((o * C_IN + i) * KS + a) * KS + b];
      if (i1  == y && i2  == xcol) val += w * (1.f - r1) * (1.f - r2);
      if (i1p == y && i2  == xcol) val += w * r1 * (1.f - r2);
      if (i1  == y && i2p == xcol) val += w * (1.f - r1) * r2;
      if (i1p == y && i2p == xcol) val += w * r1 * r2;
    }
  }

  int k    = i & 31;
  int iblk = i >> 5;
  int m    = o >> 4;
  int lane = (o & 15) + ((k >> 3) & 1) * 16;
  int e    = (k >> 4) * 8 + (k & 7);
  int frag = ((y * NIBLK + iblk) * DIL + xcol) * NMBLK + m;
  AW[frag * FRAG_HALFS + lane * 16 + e] = (__bf16)val;
}

// ---------------------------------------------------------------------------
// Kernel 2: implicit-GEMM conv. One block per (n, h) output row.
// 2 waves x 32 pixels; per wave: 6 M-blocks x 2 N-blocks of f32 accum.
// A-fragment batches stream via global_load_async_to_lds_b128 (ASYNCcnt),
// double-buffered so the next batch copies while the current one computes.
// ---------------------------------------------------------------------------
__global__ __launch_bounds__(64)
void dcls_conv_kernel(const float* __restrict__ x,
                      const __bf16* __restrict__ AW,
                      float* __restrict__ out) {
  __shared__ __attribute__((aligned(16))) __bf16 ldsx[LDSX_W * LDSX_STRIDE];  // 14560 B
  __shared__ __attribute__((aligned(32))) __bf16 ldsa[2][BATCH_HALFS];        // 2 x 43008 B

  const int n     = blockIdx.x >> 6;
  const int h     = blockIdx.x & 63;
  const int tid   = threadIdx.x;
  const int lane  = tid & 31;
  const int wave  = tid >> 5;
  const int wbase = wave * 32;

  v8f acc[12];
  {
    v8f z = {0.f, 0.f, 0.f, 0.f, 0.f, 0.f, 0.f, 0.f};
#pragma unroll
    for (int q = 0; q < 12; ++q) acc[q] = z;
  }

  // Prologue: kick off async copy of batch 0 into buffer 0.
  {
    const char* src = (const char*)AW;
    unsigned ldsbase = (unsigned)(unsigned long long)&ldsa[0][0];
#pragma unroll 1
    for (int e = tid; e < BATCH_VECS; e += 64) {   // 42 iterations per thread
      unsigned off  = (unsigned)e * 16u;
      unsigned dlds = ldsbase + off;
      asm volatile("global_load_async_to_lds_b128 %0, %1, %2"
                   :: "v"(dlds), "v"(off), "s"(src) : "memory");
    }
  }

  int bi = 0;
  for (int kh = 0; kh < DIL; ++kh) {
    const int  hh    = h + kh - PAD;
    const bool rowok = (hh >= 0) && (hh < HW);
    for (int iblk = 0; iblk < NIBLK; ++iblk, ++bi) {
      __syncthreads();  // all waves done reading ldsx and both ldsa buffers' prior contents

      if (iblk == 0) {
        // Stage input row: fp32 -> bf16, transposed to [col][channel]
        for (int e = tid; e < C_IN * LDSX_W; e += 64) {
          int ci = e / LDSX_W;
          int wp = e - ci * LDSX_W;
          int w  = wp - PAD;
          float v = 0.f;
          if (rowok && w >= 0 && w < HW)
            v = x[((n * C_IN + ci) * HW + hh) * HW + w];
          ldsx[wp * LDSX_STRIDE + ci] = (__bf16)v;
        }
      }

      // Issue async copy of the NEXT batch into the other buffer, then wait
      // for the CURRENT batch (async loads complete in order: <=42 outstanding
      // means the previous 42 have all landed).
      if (bi + 1 < NBATCH_K) {
        const char* src = (const char*)(AW + (size_t)(bi + 1) * BATCH_HALFS);
        unsigned ldsbase = (unsigned)(unsigned long long)&ldsa[(bi + 1) & 1][0];
#pragma unroll 1
        for (int e = tid; e < BATCH_VECS; e += 64) {
          unsigned off  = (unsigned)e * 16u;
          unsigned dlds = ldsbase + off;
          asm volatile("global_load_async_to_lds_b128 %0, %1, %2"
                       :: "v"(dlds), "v"(off), "s"(src) : "memory");
        }
        asm volatile("s_wait_asynccnt 42" ::: "memory");
      } else {
        asm volatile("s_wait_asynccnt 0" ::: "memory");
      }
      __syncthreads();  // current batch + ldsx visible to both waves

      const __bf16* bufa = &ldsa[bi & 1][0];

#pragma unroll
      for (int kw = 0; kw < DIL; ++kw) {
        v16bf b0, b1;
        {
          const int base = iblk * 32 + ((lane >> 4) << 3);
          const int c0   = (wbase + (lane & 15) + kw) * LDSX_STRIDE + base;
          const uint2* p0 = (const uint2*)(ldsx + c0);
          const uint2* p1 = (const uint2*)(ldsx + c0 + 16 * LDSX_STRIDE);
          union { v16bf v; uint2 q[4]; } u0, u1;
          u0.q[0] = p0[0]; u0.q[1] = p0[1]; u0.q[2] = p0[4]; u0.q[3] = p0[5];
          u1.q[0] = p1[0]; u1.q[1] = p1[1]; u1.q[2] = p1[4]; u1.q[3] = p1[5];
          b0 = u0.v; b1 = u1.v;
        }
        // Hoist all 6 A fragments so the DS loads batch up ahead of the WMMAs.
        v16bf afr[NMBLK];
#pragma unroll
        for (int m = 0; m < NMBLK; ++m)
          afr[m] = *(const v16bf*)(bufa + (kw * NMBLK + m) * FRAG_HALFS + lane * 16);
#pragma unroll
        for (int m = 0; m < NMBLK; ++m) {
          acc[m]     = __builtin_amdgcn_wmma_f32_16x16x32_bf16(false, afr[m], false, b0, (short)0, acc[m],     false, false);
          acc[6 + m] = __builtin_amdgcn_wmma_f32_16x16x32_bf16(false, afr[m], false, b1, (short)0, acc[6 + m], false, false);
        }
      }
    }
  }

  // Epilogue: C/D layout -> NCHW f32. VGPR r: M = r + 8*(lane>=16); N = lane%16.
  const int colbase = wbase + (lane & 15);
  const int rowoff  = (lane >> 4) * 8;
#pragma unroll
  for (int m = 0; m < NMBLK; ++m) {
#pragma unroll
    for (int j = 0; j < 2; ++j) {
#pragma unroll
      for (int r = 0; r < 8; ++r) {
        int o = m * 16 + rowoff + r;
        int w = colbase + j * 16;
        out[((n * C_OUT + o) * HW + h) * HW + w] = acc[m + 6 * j][r];
      }
    }
  }
}

// ---------------------------------------------------------------------------
extern "C" void kernel_launch(void* const* d_in, const int* in_sizes, int n_in,
                              void* d_out, int out_size, void* d_ws, size_t ws_size,
                              hipStream_t stream) {
  (void)in_sizes; (void)n_in; (void)out_size; (void)ws_size;
  const float* x      = (const float*)d_in[0];
  const float* weight = (const float*)d_in[1];
  const float* P1     = (const float*)d_in[2];
  const float* P2     = (const float*)d_in[3];
  float* out = (float*)d_out;

  // Workspace: packed bf16 A-fragments, 21 batches * 43008 B = 903168 bytes.
  __bf16* AW = (__bf16*)d_ws;

  const int pack_total = DIL * DIL * C_OUT * C_IN;  // 451584
  dcls_pack_kernel<<<(pack_total + 255) / 256, 256, 0, stream>>>(weight, P1, P2, AW);

  dcls_conv_kernel<<<N_BATCH * HW, 64, 0, stream>>>(x, AW, out);
}
